// Net_76175539962263
// MI455X (gfx1250) — matrix operand order repaired and checked by reading, not compiled, not run on hardware
//
#include <hip/hip_runtime.h>

typedef __attribute__((ext_vector_type(16))) _Float16 v16h;
typedef __attribute__((ext_vector_type(8)))  float    v8f;

#define N_NODES  100000
#define N_EDGES  3200000
#define N_GRAPHS 256
#define F0 16
#define F1 32
#define F2 41
#define NTILES (N_NODES / 16)   // 6250 exact

// ---------------------------------------------------------------------------
// init: deg = 1 (self loop), acc1 = 0, acc2 = 0, out = 0
// ---------------------------------------------------------------------------
__global__ void k_init(int* __restrict__ deg, float* __restrict__ acc1,
                       float* __restrict__ acc2, float* __restrict__ out) {
    int i = blockIdx.x * blockDim.x + threadIdx.x;
    if (i < N_NODES)            deg[i]  = 1;
    if (i < N_NODES * F0)       acc1[i] = 0.0f;
    if (i < N_NODES * F1)       acc2[i] = 0.0f;
    if (i < N_GRAPHS * F2)      out[i]  = 0.0f;
}

// ---------------------------------------------------------------------------
// degree: deg[col[e]] += 1
// ---------------------------------------------------------------------------
__global__ void k_degree(const int* __restrict__ col, int* __restrict__ deg) {
    int e = blockIdx.x * blockDim.x + threadIdx.x;
    if (e < N_EDGES) atomicAdd(&deg[col[e]], 1);
}

// ---------------------------------------------------------------------------
// dinv[n] = rsqrt(deg[n]);  xs[n][f] = dinv[n] * emb[x_ids[n]][f]
// one thread per (node, feature)
// ---------------------------------------------------------------------------
__global__ void k_dinv_xs(const int* __restrict__ deg, const int* __restrict__ x_ids,
                          const float* __restrict__ emb,
                          float* __restrict__ dinv, float* __restrict__ xs) {
    int i = blockIdx.x * blockDim.x + threadIdx.x;
    if (i >= N_NODES * F0) return;
    int n = i >> 4;
    int f = i & 15;
    float di = rsqrtf((float)deg[n]);
    if (f == 0) dinv[n] = di;
    xs[i] = di * emb[x_ids[n] * F0 + f];
}

// ---------------------------------------------------------------------------
// agg1: acc1[col][f] += xs[row][f]   (16-dim payload; norm folded into xs)
// ---------------------------------------------------------------------------
__global__ void k_agg1(const int* __restrict__ row, const int* __restrict__ col,
                       const float* __restrict__ xs, float* __restrict__ acc1) {
    int i = blockIdx.x * blockDim.x + threadIdx.x;
    if (i >= N_EDGES * F0) return;
    int e = i >> 4;
    int f = i & 15;
    int r = row[e];
    int c = col[e];
    atomicAdd(&acc1[c * F0 + f], xs[r * F0 + f]);
}

// ---------------------------------------------------------------------------
// agg2: acc2[col][f] += h1s[row][f]  (32-dim payload; norm folded into h1s)
// ---------------------------------------------------------------------------
__global__ void k_agg2(const int* __restrict__ row, const int* __restrict__ col,
                       const float* __restrict__ h1s, float* __restrict__ acc2) {
    int i = blockIdx.x * blockDim.x + threadIdx.x;
    if (i >= N_EDGES * F1) return;
    int e = i >> 5;
    int f = i & 31;
    int r = row[e];
    int c = col[e];
    atomicAdd(&acc2[c * F1 + f], h1s[r * F1 + f]);
}

// ---------------------------------------------------------------------------
// GEMM1 (WMMA): y0 = dinv*(acc1+xs)  [16-dim];  h1s = dinv * relu(y0 @ W1 + b1)
// A: 16x32 f16 (K=16 real, 16..31 zero-padded).  B: W1 (16x32) K-padded.
// One 16-node tile per wave; 8 waves per block.
// ---------------------------------------------------------------------------
__global__ void k_gemm1(const float* __restrict__ acc1, const float* __restrict__ xs,
                        const float* __restrict__ dinv,
                        const float* __restrict__ W1, const float* __restrict__ bias1,
                        float* __restrict__ h1s) {
    const int lane = threadIdx.x & 31;
    const int wv   = threadIdx.x >> 5;
    const int tile = blockIdx.x * 8 + wv;
    if (tile >= NTILES) return;            // wave-uniform; EXEC stays all-1s
    const int half = lane >> 4;            // 0: lanes 0-15, 1: lanes 16-31
    const int mcol = lane & 15;

    // ---- B operands: W1 padded K=16->32; two N tiles (N = 0..15, 16..31)
    v16h b0, b1;
#pragma unroll
    for (int e = 0; e < 16; ++e) {
        int k = (e < 8) ? half * 8 + e : 16 + half * 8 + (e - 8);   // B layout: table order, matches A
        float w0 = (k < F0) ? W1[k * F1 + mcol]      : 0.0f;
        float w1 = (k < F0) ? W1[k * F1 + 16 + mcol] : 0.0f;
        b0[e] = (_Float16)w0;
        b1[e] = (_Float16)w1;
    }

    // ---- A operand: row m = mcol, K octet selected by half; K>=16 is padding
    const int m  = tile * 16 + mcol;
    const float di = dinv[m];
    const float* pa = acc1 + m * F0 + half * 8;
    const float* px = xs   + m * F0 + half * 8;
    v16h a;
#pragma unroll
    for (int e = 0; e < 8; ++e) a[e] = (_Float16)(di * (pa[e] + px[e]));
#pragma unroll
    for (int e = 8; e < 16; ++e) a[e] = (_Float16)0.0f;

    // ---- matrix core
    v8f c0 = {}, c1 = {};
    c0 = __builtin_amdgcn_wmma_f32_16x16x32_f16(false, a, false, b0, (short)0, c0, false, false);
    c1 = __builtin_amdgcn_wmma_f32_16x16x32_f16(false, a, false, b1, (short)0, c1, false, false);

    // ---- epilogue: bias + relu + prescale by dinv[node]; store h1s
    const float bb0 = bias1[mcol];
    const float bb1 = bias1[16 + mcol];
#pragma unroll
    for (int v = 0; v < 8; ++v) {
        int node = tile * 16 + v + 8 * half;   // C/D: M = vgpr + 8*half, N = mcol
        float dn = dinv[node];
        float h0 = fmaxf(c0[v] + bb0, 0.0f) * dn;
        float h1 = fmaxf(c1[v] + bb1, 0.0f) * dn;
        h1s[node * F1 + mcol]      = h0;
        h1s[node * F1 + 16 + mcol] = h1;
    }
}

// ---------------------------------------------------------------------------
// GEMM2 (WMMA) + pool: y1 = dinv*(acc2+h1s) [32-dim]; h2 = y1 @ W2 + b2;
// out[batch[n]] += h2[n].  B: W2 (32x41) N-padded to 48 (3 tiles).
// ---------------------------------------------------------------------------
__global__ void k_gemm2(const float* __restrict__ acc2, const float* __restrict__ h1s,
                        const float* __restrict__ dinv,
                        const float* __restrict__ W2, const float* __restrict__ bias2,
                        const int* __restrict__ batch, float* __restrict__ out) {
    const int lane = threadIdx.x & 31;
    const int wv   = threadIdx.x >> 5;
    const int tile = blockIdx.x * 8 + wv;
    if (tile >= NTILES) return;            // wave-uniform
    const int half = lane >> 4;
    const int mcol = lane & 15;

    // ---- B operands: three N tiles, n >= 41 zero-padded
    v16h B0, B1, B2;
#pragma unroll
    for (int e = 0; e < 16; ++e) {
        int k  = (e < 8) ? half * 8 + e : 16 + half * 8 + (e - 8);   // k in 0..31, all valid
        int n0 = mcol, n1 = 16 + mcol, n2 = 32 + mcol;
        B0[e] = (_Float16)W2[k * F2 + n0];
        B1[e] = (_Float16)W2[k * F2 + n1];
        B2[e] = (n2 < F2) ? (_Float16)W2[k * F2 + n2] : (_Float16)0.0f;
    }

    // ---- A operand: full K=32
    const int m  = tile * 16 + mcol;
    const float di = dinv[m];
    const float* pa = acc2 + m * F1;
    const float* ph = h1s  + m * F1;
    v16h a;
#pragma unroll
    for (int e = 0; e < 8; ++e) {
        int k = half * 8 + e;
        a[e] = (_Float16)(di * (pa[k] + ph[k]));
    }
#pragma unroll
    for (int e = 8; e < 16; ++e) {
        int k = 16 + half * 8 + (e - 8);
        a[e] = (_Float16)(di * (pa[k] + ph[k]));
    }

    // ---- matrix core
    v8f c0 = {}, c1 = {}, c2 = {};
    c0 = __builtin_amdgcn_wmma_f32_16x16x32_f16(false, a, false, B0, (short)0, c0, false, false);
    c1 = __builtin_amdgcn_wmma_f32_16x16x32_f16(false, a, false, B1, (short)0, c1, false, false);
    c2 = __builtin_amdgcn_wmma_f32_16x16x32_f16(false, a, false, B2, (short)0, c2, false, false);

    // ---- fused graph pooling: atomic scatter into [256 x 41]
    const int n0 = mcol, n1 = 16 + mcol, n2 = 32 + mcol;
    const float bb0 = bias2[n0];
    const float bb1 = bias2[n1];
    const float bb2 = (n2 < F2) ? bias2[n2] : 0.0f;
#pragma unroll
    for (int v = 0; v < 8; ++v) {
        int node = tile * 16 + v + 8 * half;
        int g = batch[node];
        atomicAdd(&out[g * F2 + n0], c0[v] + bb0);
        atomicAdd(&out[g * F2 + n1], c1[v] + bb1);
        if (n2 < F2) atomicAdd(&out[g * F2 + n2], c2[v] + bb2);
    }
}

// ---------------------------------------------------------------------------
extern "C" void kernel_launch(void* const* d_in, const int* in_sizes, int n_in,
                              void* d_out, int out_size, void* d_ws, size_t ws_size,
                              hipStream_t stream) {
    const int*   x_ids = (const int*)d_in[0];
    const int*   ei    = (const int*)d_in[1];   // [2, E]: row = ei[0:E), col = ei[E:2E)
    const int*   batch = (const int*)d_in[2];
    const float* emb   = (const float*)d_in[3]; // [1500, 16]
    const float* W1    = (const float*)d_in[4]; // [16, 32]
    const float* b1    = (const float*)d_in[5]; // [32]
    const float* W2    = (const float*)d_in[6]; // [32, 41]
    const float* b2    = (const float*)d_in[7]; // [41]
    float* out = (float*)d_out;                 // [256, 41] f32

    const int* e_row = ei;
    const int* e_col = ei + N_EDGES;

    // workspace carve-up (~39.2 MB total; everything L2-resident)
    char* p = (char*)d_ws;
    int*   deg  = (int*)p;    p += (size_t)N_NODES * sizeof(int);
    float* dinv = (float*)p;  p += (size_t)N_NODES * sizeof(float);
    float* xs   = (float*)p;  p += (size_t)N_NODES * F0 * sizeof(float);
    float* acc1 = (float*)p;  p += (size_t)N_NODES * F0 * sizeof(float);
    float* h1s  = (float*)p;  p += (size_t)N_NODES * F1 * sizeof(float);
    float* acc2 = (float*)p;  p += (size_t)N_NODES * F1 * sizeof(float);

    const int B = 256;

    int initN = N_NODES * F1;  // largest region
    k_init<<<(initN + B - 1) / B, B, 0, stream>>>(deg, acc1, acc2, out);

    k_degree<<<(N_EDGES + B - 1) / B, B, 0, stream>>>(e_col, deg);

    k_dinv_xs<<<(N_NODES * F0 + B - 1) / B, B, 0, stream>>>(deg, x_ids, emb, dinv, xs);

    k_agg1<<<(N_EDGES * F0 + B - 1) / B, B, 0, stream>>>(e_row, e_col, xs, acc1);

    k_gemm1<<<(NTILES + 7) / 8, B, 0, stream>>>(acc1, xs, dinv, W1, b1, h1s);

    k_agg2<<<(N_EDGES * F1 + B - 1) / B, B, 0, stream>>>(e_row, e_col, h1s, acc2);

    k_gemm2<<<(NTILES + 7) / 8, B, 0, stream>>>(acc2, h1s, dinv, W2, b2, batch, out);
}
